// MultiHeadAttention_81664508166458
// MI455X (gfx1250) — matrix operand order repaired and checked by reading, not compiled, not run on hardware
//
#include <hip/hip_runtime.h>
#include <hip/hip_bf16.h>

typedef __attribute__((ext_vector_type(16))) __bf16 v16bf;
typedef __attribute__((ext_vector_type(8)))  __bf16 v8bf;
typedef __attribute__((ext_vector_type(8)))  float  v8f;

#define BM 128
#define BN 128
#define BK 32
#define LDT 40   // padded LDS stride in bf16 elems (80B) -> 16B aligned rows, broken bank cycles

// ---------------------------------------------------------------------------
// Tiled NT GEMM: C[m,n] = sum_k A[m,k] * W[n,k] + bias[n]
// A: fp32 or bf16 (template), W/bias: fp32, C: bf16 or fp32 (template).
// 256 threads = 8 waves; wave tile 32x64 => 8x v_wmma_f32_16x16x32_bf16 / K-step.
// Double-buffered LDS, register-staged global loads issued a full tile ahead.
// ---------------------------------------------------------------------------
template<bool A_IS_BF16, bool OUT_IS_BF16>
__global__ __launch_bounds__(256)
void gemm_nt_wmma(const void* __restrict__ Avoid,
                  const float* __restrict__ W,
                  const float* __restrict__ bias,
                  void* __restrict__ Cvoid,
                  int M, int N, int K)
{
    __shared__ __align__(16) __bf16 As[2][BM * LDT];
    __shared__ __align__(16) __bf16 Bs[2][BN * LDT];

    const int tid   = threadIdx.x;
    const int lane  = tid & 31;
    const int wave  = tid >> 5;       // 0..7
    const int wm    = wave & 3;       // 4 row-groups of 32
    const int wn    = wave >> 2;      // 2 col-groups of 64
    const int lrow  = lane & 15;
    const int khalf = lane >> 4;      // 0/1

    const int m0 = blockIdx.y * BM;
    const int n0 = blockIdx.x * BN;

    const float*  Af = (const float*)Avoid;
    const __bf16* Ab = (const __bf16*)Avoid;

    // Staging coordinates: 512 "8-elem" units over a 128x32 tile; thread owns 2.
    int srow[2], scol[2];
    #pragma unroll
    for (int i = 0; i < 2; ++i) {
        int idx = tid + i * 256;
        srow[i] = idx >> 2;           // 0..127
        scol[i] = (idx & 3) * 8;      // 0,8,16,24
    }

    v8f acc[2][4] = {};

    float4 ar[4], br[4];              // fp32 staging (A and B)
    v8bf   ar8[2];                    // bf16-A staging

    // ---- issue global loads for K-tile kt into staging registers ----
    auto issue_loads = [&](int kt) {
        #pragma unroll
        for (int i = 0; i < 2; ++i) {
            if constexpr (A_IS_BF16) {
                ar8[i] = *(const v8bf*)(Ab + (size_t)(m0 + srow[i]) * K + kt + scol[i]);
            } else {
                const float* sa = Af + (size_t)(m0 + srow[i]) * K + kt + scol[i];
                ar[2 * i]     = *(const float4*)sa;
                ar[2 * i + 1] = *(const float4*)(sa + 4);
            }
            const float* sb = W + (size_t)(n0 + srow[i]) * K + kt + scol[i];
            br[2 * i]     = *(const float4*)sb;
            br[2 * i + 1] = *(const float4*)(sb + 4);
        }
    };

    // ---- convert staged registers and store to LDS buffer p ----
    auto store_lds = [&](int p) {
        #pragma unroll
        for (int i = 0; i < 2; ++i) {
            if constexpr (A_IS_BF16) {
                *(v8bf*)&As[p][srow[i] * LDT + scol[i]] = ar8[i];
            } else {
                v8bf pk;
                pk[0] = (__bf16)ar[2*i].x; pk[1] = (__bf16)ar[2*i].y;
                pk[2] = (__bf16)ar[2*i].z; pk[3] = (__bf16)ar[2*i].w;
                pk[4] = (__bf16)ar[2*i+1].x; pk[5] = (__bf16)ar[2*i+1].y;
                pk[6] = (__bf16)ar[2*i+1].z; pk[7] = (__bf16)ar[2*i+1].w;
                *(v8bf*)&As[p][srow[i] * LDT + scol[i]] = pk;
            }
            v8bf pb;
            pb[0] = (__bf16)br[2*i].x; pb[1] = (__bf16)br[2*i].y;
            pb[2] = (__bf16)br[2*i].z; pb[3] = (__bf16)br[2*i].w;
            pb[4] = (__bf16)br[2*i+1].x; pb[5] = (__bf16)br[2*i+1].y;
            pb[6] = (__bf16)br[2*i+1].z; pb[7] = (__bf16)br[2*i+1].w;
            *(v8bf*)&Bs[p][srow[i] * LDT + scol[i]] = pb;
        }
    };

    issue_loads(0);                   // prime the pipeline
    int p = 0;
    for (int kt = 0; kt < K; kt += BK) {
        store_lds(p);                 // waits on in-flight loads, cvt_pk, ds_store_b128
        __syncthreads();              // implicit s_wait_dscnt 0 -> double-buffer safe

        if (kt + BK < K)
            issue_loads(kt + BK);     // next tile's 8 b128 loads fly during WMMA

        if (kt + 2 * BK < K) {        // uniform-guarded L2 prefetch two tiles ahead
            if constexpr (A_IS_BF16)
                __builtin_prefetch(Ab + (size_t)(m0 + srow[0]) * K + kt + 2 * BK + scol[0], 0, 3);
            else
                __builtin_prefetch(Af + (size_t)(m0 + srow[0]) * K + kt + 2 * BK + scol[0], 0, 3);
            __builtin_prefetch(W + (size_t)(n0 + srow[0]) * K + kt + 2 * BK + scol[0], 0, 3);
        }

        // ---- fragments per ISA VGPR layouts ----
        const __bf16* Asp = As[p];
        const __bf16* Bsp = Bs[p];
        v16bf afrag[2], bfrag[4];
        #pragma unroll
        for (int mi = 0; mi < 2; ++mi) {
            int r = wm * 32 + mi * 16 + lrow;
            // A 16x32: lane<16 holds K{0..7,16..23}; lane>=16 holds K{8..15,24..31}
            v8bf lo = *(const v8bf*)&Asp[r * LDT + khalf * 8];
            v8bf hi = *(const v8bf*)&Asp[r * LDT + 16 + khalf * 8];
            afrag[mi] = __builtin_shufflevector(lo, hi,
                0,1,2,3,4,5,6,7,8,9,10,11,12,13,14,15);
        }
        #pragma unroll
        for (int ni = 0; ni < 4; ++ni) {
            int r = wn * 64 + ni * 16 + lrow;
            // B 32x16: lane<16 holds K{0..7,16..23}; lane>=16 holds K{8..15,24..31}
            v8bf lo = *(const v8bf*)&Bsp[r * LDT + khalf * 8];
            v8bf hi = *(const v8bf*)&Bsp[r * LDT + 16 + khalf * 8];
            bfrag[ni] = __builtin_shufflevector(lo, hi,
                0,1,2,3,4,5,6,7,8,9,10,11,12,13,14,15);
        }

        #pragma unroll
        for (int mi = 0; mi < 2; ++mi)
            #pragma unroll
            for (int ni = 0; ni < 4; ++ni)
                acc[mi][ni] = __builtin_amdgcn_wmma_f32_16x16x32_bf16(
                    false, afrag[mi], false, bfrag[ni],
                    (short)0, acc[mi][ni], false, false);

        p ^= 1;
    }

    // ---- epilogue: bias + store (C layout: lane<16 -> M=r, lane>=16 -> M=r+8; N=lane&15) ----
    float*  Cf = (float*)Cvoid;
    __bf16* Cb = (__bf16*)Cvoid;
    #pragma unroll
    for (int ni = 0; ni < 4; ++ni) {
        int n = n0 + wn * 64 + ni * 16 + lrow;
        float bv = bias[n];
        #pragma unroll
        for (int mi = 0; mi < 2; ++mi) {
            int mbase = m0 + wm * 32 + mi * 16 + khalf * 8;
            #pragma unroll
            for (int r = 0; r < 8; ++r) {
                float val = acc[mi][ni][r] + bv;
                size_t off = (size_t)(mbase + r) * N + n;
                if constexpr (OUT_IS_BF16) Cb[off] = (__bf16)val;
                else                       Cf[off] = val;
            }
        }
    }
}

// ---------------------------------------------------------------------------
// Per-position across-head attention with fused RoPE.
// One block (256 thr) per (b,s) position: q,k,v are [16 heads x 128 dim] bf16.
// qk[h,t] = <q_h, k_t>/sqrt(128); softmax over t; out[h,d] = sum_t att*v.
// Output written pre-scrambled as the A-matrix of the final GEMM.
// ---------------------------------------------------------------------------
#define SEQ 2048
#define NH  16
#define HD  128
#define EMB 2048

__global__ __launch_bounds__(256)
void attn_rope_kernel(const __bf16* __restrict__ q,
                      const __bf16* __restrict__ k,
                      const __bf16* __restrict__ v,
                      __bf16* __restrict__ ao)
{
    __shared__ float qs[NH][HD + 4];
    __shared__ float ks[NH][HD + 4];
    __shared__ float vs[NH][HD + 4];
    __shared__ float qks[NH][NH];

    const int pos = blockIdx.x;           // b*S + s
    const int s   = pos & (SEQ - 1);
    const size_t rowoff = (size_t)pos * EMB;
    const float log_theta_over_d = 9.2103403719761836f / (float)HD; // ln(10000)/128

    // stage q,k with RoPE: pairs (d, d+1), angle = s * theta^(-d/HD)
    for (int idx = threadIdx.x; idx < NH * HD / 2; idx += 256) {
        int h = idx >> 6;                 // /(HD/2)
        int i = idx & 63;
        int d = 2 * i;
        float ang = (float)s * __expf(-(float)d * log_theta_over_d);
        float sn, cs;
        __sincosf(ang, &sn, &cs);
        float x0 = (float)q[rowoff + h * HD + d];
        float x1 = (float)q[rowoff + h * HD + d + 1];
        qs[h][d]     = x0 * cs - x1 * sn;
        qs[h][d + 1] = x0 * sn + x1 * cs;
        x0 = (float)k[rowoff + h * HD + d];
        x1 = (float)k[rowoff + h * HD + d + 1];
        ks[h][d]     = x0 * cs - x1 * sn;
        ks[h][d + 1] = x0 * sn + x1 * cs;
    }
    for (int idx = threadIdx.x; idx < NH * HD; idx += 256)
        vs[idx >> 7][idx & (HD - 1)] = (float)v[rowoff + idx];
    __syncthreads();

    // 16x16 scores: thread (h,t)
    {
        int h = threadIdx.x >> 4, t = threadIdx.x & 15;
        float a = 0.f;
        #pragma unroll 8
        for (int d = 0; d < HD; ++d) a += qs[h][d] * ks[t][d];
        qks[h][t] = a * 0.08838834764831845f;   // 1/sqrt(128)
    }
    __syncthreads();

    // row softmax (16 rows, one thread each; trivial work)
    if (threadIdx.x < NH) {
        int h = threadIdx.x;
        float mx = qks[h][0];
        #pragma unroll
        for (int t = 1; t < NH; ++t) mx = fmaxf(mx, qks[h][t]);
        float sum = 0.f;
        #pragma unroll
        for (int t = 0; t < NH; ++t) { float e = __expf(qks[h][t] - mx); qks[h][t] = e; sum += e; }
        float r = 1.f / sum;
        #pragma unroll
        for (int t = 0; t < NH; ++t) qks[h][t] *= r;
    }
    __syncthreads();

    // out[h,d] = sum_t att[h,t]*v[t,d]; store "scrambled" = row-major A for final GEMM:
    // flat index within batch = h*S*D + s*D + d  (== reference swapaxes+reshape)
    const size_t b = (size_t)(pos >> 11);     // / SEQ
    for (int idx = threadIdx.x; idx < NH * HD; idx += 256) {
        int h = idx >> 7, d = idx & (HD - 1);
        float o = 0.f;
        #pragma unroll
        for (int t = 0; t < NH; ++t) o += qks[h][t] * vs[t][d];
        size_t dst = b * ((size_t)SEQ * EMB) + (size_t)h * SEQ * HD + (size_t)s * HD + d;
        ao[dst] = (__bf16)o;
    }
}

// ---------------------------------------------------------------------------
extern "C" void kernel_launch(void* const* d_in, const int* in_sizes, int n_in,
                              void* d_out, int out_size, void* d_ws, size_t ws_size,
                              hipStream_t stream) {
    (void)in_sizes; (void)n_in; (void)out_size; (void)ws_size;
    const float* x  = (const float*)d_in[0];
    const float* Wq = (const float*)d_in[1];
    const float* bq = (const float*)d_in[2];
    const float* Wk = (const float*)d_in[3];
    const float* bk = (const float*)d_in[4];
    const float* Wv = (const float*)d_in[5];
    const float* bv = (const float*)d_in[6];
    const float* Wo = (const float*)d_in[7];
    const float* bo = (const float*)d_in[8];

    const int Bb = 4, S = SEQ, E = EMB;
    const int Mrows = Bb * S;                 // 8192
    const size_t elems = (size_t)Mrows * E;   // 16.7M

    __bf16* qws = (__bf16*)d_ws;
    __bf16* kws = qws + elems;
    __bf16* vws = kws + elems;
    __bf16* aow = vws + elems;                // 4 x 32MB bf16 = 128MB workspace

    dim3 grid(E / BN, Mrows / BM);            // (16, 64)
    gemm_nt_wmma<false, true><<<grid, 256, 0, stream>>>(x, Wq, bq, qws, Mrows, E, E);
    gemm_nt_wmma<false, true><<<grid, 256, 0, stream>>>(x, Wk, bk, kws, Mrows, E, E);
    gemm_nt_wmma<false, true><<<grid, 256, 0, stream>>>(x, Wv, bv, vws, Mrows, E, E);

    attn_rope_kernel<<<Mrows, 256, 0, stream>>>(qws, kws, vws, aow);

    gemm_nt_wmma<true, false><<<grid, 256, 0, stream>>>(aow, Wo, bo, d_out, Mrows, E, E);
}
